// GraphAttentionLayer_28776280883394
// MI455X (gfx1250) — compile-verified, matrix operand order checked by reading, not run on hardware
//
#include <hip/hip_runtime.h>
#include <hip/hip_bf16.h>

// Fused GAT layer for MI455X (gfx1250, wave32, WMMA).
//
// Roofline: mandatory traffic = one pass over adj (256 MB) ~ 11us @ 23.3 TB/s.
// Flash-style single pass; per-row softmax shift uses the provable bound
// M_i = max(0, s1[i] + max_j s2[j]) so exp-weights are <= 1, no online
// rescaling, and split-K partials combine by plain addition.
// Scores are kept in the log2 domain (s1,s2 pre-scaled by log2(e)) and the
// edge weight is a single raw v_exp_f32 (__builtin_amdgcn_exp2f — no libm
// denormal fixup; args are <= 0 and flush-to-zero is desired).
// The att@Wh MACs run on v_wmma_f32_16x16x32_f16; the softmax denominator is
// a 5th WMMA against an all-ones B (replaces a 16-deep serial add chain).
// K-split count chosen at launch from ws_size (more waves -> better latency
// hiding for the load->wmma waits the backend exposes).

typedef __attribute__((ext_vector_type(16))) _Float16 v16h;
typedef __attribute__((ext_vector_type(8)))  _Float16 v8h;
typedef __attribute__((ext_vector_type(8)))  float    v8f;

#define NN      8192
#define INF_    128
#define OUTF    64
#define NEG_SLOPE 0.2f
#define LOG2E   1.4426950408889634f

// ---------------------------------------------------------------------------
// Kernel 1: Wh = x@W (stored transposed as f16 for WMMA B fragments),
//           s1 = (Wh@a1)*log2e, s2 = (Wh@a2)*log2e.  (134 MFLOP — negligible)
// ---------------------------------------------------------------------------
__global__ void __launch_bounds__(64)
wh_kernel(const float* __restrict__ x, const float* __restrict__ W,
          const float* __restrict__ a,
          _Float16* __restrict__ whT, float* __restrict__ s1, float* __restrict__ s2)
{
    const int row = blockIdx.x;          // 0..8191
    const int n   = threadIdx.x;         // 0..63
    const float* xr = x + (size_t)row * INF_;
    float acc = 0.f;
#pragma unroll 8
    for (int k = 0; k < INF_; ++k)
        acc = fmaf(xr[k], W[k * OUTF + n], acc);

    whT[(size_t)n * NN + row] = (_Float16)acc;   // transposed f16 copy

    __shared__ float sm1[OUTF], sm2[OUTF];
    sm1[n] = acc * a[n];
    sm2[n] = acc * a[OUTF + n];
    __syncthreads();
    for (int off = OUTF / 2; off > 0; off >>= 1) {
        if (n < off) { sm1[n] += sm1[n + off]; sm2[n] += sm2[n + off]; }
        __syncthreads();
    }
    if (n == 0) { s1[row] = sm1[0] * LOG2E; s2[row] = sm2[0] * LOG2E; }
}

// ---------------------------------------------------------------------------
// Kernel 2: S2MAX = max_j s2[j]   (single block reduction over 8192 values)
// ---------------------------------------------------------------------------
__global__ void __launch_bounds__(256)
s2max_kernel(const float* __restrict__ s2, float* __restrict__ s2max)
{
    __shared__ float sm[256];
    float m = -3.0e38f;
    for (int i = threadIdx.x; i < NN; i += 256) m = fmaxf(m, s2[i]);
    sm[threadIdx.x] = m;
    __syncthreads();
    for (int off = 128; off > 0; off >>= 1) {
        if (threadIdx.x < off) sm[threadIdx.x] = fmaxf(sm[threadIdx.x], sm[threadIdx.x + off]);
        __syncthreads();
    }
    if (threadIdx.x == 0) *s2max = sm[0];
}

// ---------------------------------------------------------------------------
// Kernel 3: fused masked-softmax-weighted aggregation via WMMA.
// One wave = one 16-row M tile x one K split (jspan columns).
// A fragment built per the CDNA5 16-bit A layout:
//   lanes 0-15 : M=lane,    K in {0..7, 16..23}
//   lanes 16-31: M=lane-16, K in {8..15, 24..31}
// B fragment mirrors it with N in place of M; whT rows make it 2x b128 loads.
// ---------------------------------------------------------------------------
__device__ __forceinline__ _Float16
edge_p(int aij, float s2v, float s1r, float Mi)
{
    float t  = s1r + s2v;                  // log2-domain score
    float lr = fmaxf(t, NEG_SLOPE * t);    // leaky_relu (scale-invariant)
    float v  = (aij > 0) ? lr : 0.0f;      // masked entries keep value 0
    // raw v_exp_f32: arg <= 0 by construction; FTZ on huge negatives is fine
    return (_Float16)__builtin_amdgcn_exp2f(v - Mi);
}

__global__ void __launch_bounds__(128)
gat_flash_kernel(const int* __restrict__ adj,
                 const float* __restrict__ s1, const float* __restrict__ s2g,
                 const float* __restrict__ s2maxp,
                 const _Float16* __restrict__ whT,
                 float* __restrict__ accp, float* __restrict__ lp,
                 int jspan)
{
    const int lane  = threadIdx.x & 31;
    const int wave  = threadIdx.x >> 5;
    const int mtile = blockIdx.x * 4 + wave;       // 0..511
    const int m0    = mtile * 16;
    const int split = blockIdx.y;
    const int jbeg  = split * jspan;
    const int jend  = jbeg + jspan;

    const int   mrow = m0 + (lane & 15);
    const int   klo  = (lane < 16) ? 0 : 8;
    const float S2MAX = *s2maxp;
    const float s1r   = s1[mrow];
    const float Mi    = fmaxf(0.0f, s1r + S2MAX); // fixed per-row shift bound

    v8f acc0 = {}, acc1 = {}, acc2 = {}, acc3 = {};
    v8f accS = {};                                 // denominator accumulator
    v16h ONES;
#pragma unroll
    for (int e = 0; e < 16; ++e) ONES[e] = (_Float16)1.0f;

    const size_t rowbase = (size_t)mrow * NN;
    const _Float16* bbase0 = whT + (size_t)(lane & 15) * NN;

    for (int j0 = jbeg; j0 < jend; j0 += 32) {
        // stream adj: prefetch two chunks ahead (global_prefetch_b8)
        __builtin_prefetch(adj + rowbase + j0 + 64 + klo, 0, 0);

        const int4*   ap = (const int4*)  (adj + rowbase + j0 + klo);
        const float4* sp = (const float4*)(s2g + j0 + klo);
        const int4   q0 = ap[0], q1 = ap[1], q2 = ap[4], q3 = ap[5];
        const float4 f0 = sp[0], f1 = sp[1], f2 = sp[4], f3 = sp[5];

        v16h A;
        A[0]  = edge_p(q0.x, f0.x, s1r, Mi);
        A[1]  = edge_p(q0.y, f0.y, s1r, Mi);
        A[2]  = edge_p(q0.z, f0.z, s1r, Mi);
        A[3]  = edge_p(q0.w, f0.w, s1r, Mi);
        A[4]  = edge_p(q1.x, f1.x, s1r, Mi);
        A[5]  = edge_p(q1.y, f1.y, s1r, Mi);
        A[6]  = edge_p(q1.z, f1.z, s1r, Mi);
        A[7]  = edge_p(q1.w, f1.w, s1r, Mi);
        A[8]  = edge_p(q2.x, f2.x, s1r, Mi);
        A[9]  = edge_p(q2.y, f2.y, s1r, Mi);
        A[10] = edge_p(q2.z, f2.z, s1r, Mi);
        A[11] = edge_p(q2.w, f2.w, s1r, Mi);
        A[12] = edge_p(q3.x, f3.x, s1r, Mi);
        A[13] = edge_p(q3.y, f3.y, s1r, Mi);
        A[14] = edge_p(q3.z, f3.z, s1r, Mi);
        A[15] = edge_p(q3.w, f3.w, s1r, Mi);

        // B fragments: whT[n][j0+klo .. +7] and [.. +16 .. +23], contiguous f16
        const _Float16* b0 = bbase0 + j0 + klo;
        v8h b0lo = *(const v8h*)(b0);
        v8h b0hi = *(const v8h*)(b0 + 16);
        const _Float16* b1 = b0 + 16 * (size_t)NN;
        v8h b1lo = *(const v8h*)(b1);
        v8h b1hi = *(const v8h*)(b1 + 16);
        const _Float16* b2 = b0 + 32 * (size_t)NN;
        v8h b2lo = *(const v8h*)(b2);
        v8h b2hi = *(const v8h*)(b2 + 16);
        const _Float16* b3 = b0 + 48 * (size_t)NN;
        v8h b3lo = *(const v8h*)(b3);
        v8h b3hi = *(const v8h*)(b3 + 16);

        v16h B0 = __builtin_shufflevector(b0lo, b0hi, 0,1,2,3,4,5,6,7,8,9,10,11,12,13,14,15);
        v16h B1 = __builtin_shufflevector(b1lo, b1hi, 0,1,2,3,4,5,6,7,8,9,10,11,12,13,14,15);
        v16h B2 = __builtin_shufflevector(b2lo, b2hi, 0,1,2,3,4,5,6,7,8,9,10,11,12,13,14,15);
        v16h B3 = __builtin_shufflevector(b3lo, b3hi, 0,1,2,3,4,5,6,7,8,9,10,11,12,13,14,15);

        acc0 = __builtin_amdgcn_wmma_f32_16x16x32_f16(false, A, false, B0,   (short)0, acc0, false, false);
        acc1 = __builtin_amdgcn_wmma_f32_16x16x32_f16(false, A, false, B1,   (short)0, acc1, false, false);
        acc2 = __builtin_amdgcn_wmma_f32_16x16x32_f16(false, A, false, B2,   (short)0, acc2, false, false);
        acc3 = __builtin_amdgcn_wmma_f32_16x16x32_f16(false, A, false, B3,   (short)0, acc3, false, false);
        // denominator: row sums of A on the matrix pipe (B = ones)
        accS = __builtin_amdgcn_wmma_f32_16x16x32_f16(false, A, false, ONES, (short)0, accS, false, false);
    }

    // C layout: VGPR r -> row m0 + r + 8*(lane>=16), col = tile*16 + (lane&15)
    const int hi  = lane >> 4;
    const int nlo = lane & 15;
    float* accout = accp + ((size_t)split * NN + m0) * OUTF;
#pragma unroll
    for (int r = 0; r < 8; ++r) {
        float* o = accout + (size_t)(r + 8 * hi) * OUTF + nlo;
        o[0]  = acc0[r];
        o[16] = acc1[r];
        o[32] = acc2[r];
        o[48] = acc3[r];
    }
    // accS columns are identical; one lane per half writes 8 row sums
    if (nlo == 0) {
#pragma unroll
        for (int r = 0; r < 8; ++r)
            lp[(size_t)split * NN + m0 + r + 8 * hi] = accS[r];
    }
}

// ---------------------------------------------------------------------------
// Kernel 4: out = (sum_s acc_s) / (sum_s l_s)
// ---------------------------------------------------------------------------
__global__ void __launch_bounds__(256)
combine_kernel(const float* __restrict__ accp, const float* __restrict__ lp,
               float* __restrict__ out, int splits)
{
    const int idx = blockIdx.x * 256 + threadIdx.x;  // 8192*64 total
    const int row = idx >> 6;
    float num = 0.f, den = 0.f;
    for (int s = 0; s < splits; ++s) {
        num += accp[(size_t)s * NN * OUTF + idx];
        den += lp[(size_t)s * NN + row];
    }
    out[idx] = num / den;
}

// ---------------------------------------------------------------------------
// Launcher. Workspace layout (bytes):
//   [0)            whT   : 64*8192 f16  = 1,048,576
//   [+1048576)     s1    : 8192 f32     = 32,768
//   [+1081344)     s2    : 8192 f32     = 32,768
//   [+1114112)     s2max : 1 f32 (padded to 256)
//   [+1114368)     lp    : splits*8192 f32       (max 8 -> 262,144)
//   [+1376512)     accp  : splits*8192*64 f32    (max 8 -> 16,777,216)
// splits chosen largest in {8,4,2,1} that fits ws_size (host-side, fixed per
// run -> deterministic).
// ---------------------------------------------------------------------------
extern "C" void kernel_launch(void* const* d_in, const int* in_sizes, int n_in,
                              void* d_out, int out_size, void* d_ws, size_t ws_size,
                              hipStream_t stream)
{
    const float* x   = (const float*)d_in[0];
    const int*   adj = (const int*)  d_in[1];
    const float* W   = (const float*)d_in[2];
    const float* a   = (const float*)d_in[3];
    float* out = (float*)d_out;

    char* ws = (char*)d_ws;
    _Float16* whT  = (_Float16*)(ws);
    float*    s1   = (float*)(ws + 1048576);
    float*    s2   = (float*)(ws + 1081344);
    float*    s2m  = (float*)(ws + 1114112);
    float*    lp   = (float*)(ws + 1114368);
    float*    accp = (float*)(ws + 1376512);

    int splits = 8;
    while (splits > 1 &&
           (size_t)1376512 + (size_t)splits * NN * OUTF * 4 > ws_size)
        splits >>= 1;
    const int jspan = NN / splits;

    wh_kernel<<<NN, 64, 0, stream>>>(x, W, a, whT, s1, s2);
    s2max_kernel<<<1, 256, 0, stream>>>(s2, s2m);
    gat_flash_kernel<<<dim3(NN / 16 / 4, splits), 128, 0, stream>>>(
        adj, s1, s2, s2m, whT, accp, lp, jspan);
    combine_kernel<<<(NN * OUTF) / 256, 256, 0, stream>>>(accp, lp, out, splits);
}
